// HierarchicalMemory_37838661878388
// MI455X (gfx1250) — matrix operand order repaired and checked by reading, not compiled, not run on hardware
//
#include <hip/hip_runtime.h>

// ---- CDNA5 (gfx1250) vector types -----------------------------------------
typedef _Float16 h4   __attribute__((ext_vector_type(4)));   // 8 B  LDS load
typedef _Float16 v16h __attribute__((ext_vector_type(16)));  // WMMA A/B frag
typedef float    v8f  __attribute__((ext_vector_type(8)));   // WMMA C/D frag
typedef float    f4   __attribute__((ext_vector_type(4)));   // 16 B global load

#define P        400
#define P4       (P / 4)          // 100 float4 per row
#define SM       404              // LDS row stride in f16 elements (808 B)
#define NT       512              // 16 wave32s per workgroup
#define PADDED   416              // h padded to 13 K-tiles of 32
#define M_BYTES  (P * SM * 2)     // 323,200 B of f16 M in LDS
#define H16_OFF  M_BYTES          // f16 h mirror  (832 B)   | overlaps diff
#define H32_OFF  (M_BYTES + 832)  // f32 h master (1,664 B)  | overlaps diff/sum
#define SUM_OFF  (M_BYTES + 1600) // phase-1 sum[] f32
#define LDS_BYTES (M_BYTES + 3200) // 326,400 <= 327,680 (320 KB per WGP)

__device__ __forceinline__ int sid(int p) {
  return (p < 100) ? 0 : (p < 200) ? 1 : (p < 280) ? 2 : (p < 340) ? 3 : 4;
}
__device__ __forceinline__ float clip1(float v) {
  return fminf(1.0f, fmaxf(-1.0f, v));
}
__device__ __forceinline__ float lrelu(float v) {
  return v < 0.0f ? 0.01f * v : v;
}

__global__ __launch_bounds__(NT, 1)
void HierarchicalMemory_37838661878388_kernel(
    const float* __restrict__ M_prev, const float* __restrict__ p_inf,
    const float* __restrict__ p_gen,  const float* __restrict__ query,
    float* __restrict__ out) {
  extern __shared__ __align__(16) char smem[];
  _Float16* Ml  = (_Float16*)smem;               // [400][404] f16
  float*    dif = (float*)(smem + H16_OFF);      // phase 1 only
  float*    sum = (float*)(smem + SUM_OFF);      // phase 1 only
  _Float16* hl  = (_Float16*)(smem + H16_OFF);   // phase 2: f16 h (B operand)
  float*    hf  = (float*)(smem + H32_OFF);      // phase 2: f32 h master

  const int tid = threadIdx.x;
  const int b   = blockIdx.x;
  const float* gM = M_prev + (size_t)b * P * P;

  // ---------- Phase 0: per-batch diff/sum vectors --------------------------
  for (int i = tid; i < P; i += NT) {
    float a = p_inf[b * P + i], c = p_gen[b * P + i];
    dif[i] = a - c;
    sum[i] = a + c;
  }
  __syncthreads();

  // ---------- Phase 1: Hebbian update, one pass over M_prev (HBM-bound) ----
  // M = clip(0.9999*M_prev + 0.5*diff[p]*sum[q], -1, 1) * hier_mask -> f16 LDS
  const f4* gM4 = (const f4*)gM;
  for (int g = tid; g < P * P4; g += NT) {
    int row = g / P4;
    int q0  = (g - row * P4) * 4;         // stream boundaries are multiples of 4
    f4 m = gM4[g];
    float d = dif[row];
    f4 s = *(const f4*)(sum + q0);
    bool on = sid(q0) >= sid(row);        // upper-block-triangular hier mask
    h4 o;
#pragma unroll
    for (int j = 0; j < 4; ++j) {
      float v = 0.9999f * m[j] + 0.5f * d * s[j];
      v = clip1(v);
      o[j] = (_Float16)(on ? v : 0.0f);
    }
    *(h4*)(Ml + row * SM + q0) = o;
  }
  __syncthreads();

  // ---------- Phase 1b: h0 = leaky_relu(clip(query)), f32 + f16 mirror -----
  for (int i = tid; i < PADDED; i += NT) {
    float x = 0.0f;
    if (i < P) x = lrelu(clip1(query[b * P + i]));
    hf[i] = x;
    hl[i] = (_Float16)x;   // zeros at 400..415 guard the K tail tile
  }
  __syncthreads();

  // ---------- Phase 2: 5 attractor iterations via v_wmma out of LDS --------
  const int  lane = tid & 31;
  const int  wv   = __builtin_amdgcn_readfirstlane((int)(tid >> 5)); // wave id
  const bool hi   = (lane & 16) != 0;
  const int  lo   = hi ? 8 : 0;            // A-fragment K sub-offset
  const int  rsub = lane & 15;             // A-fragment row within tile
  const h4   hz   = {};

  const int bounds[5] = {400, 340, 280, 200, 100};   // active rows per iter

  for (int it = 0; it < 5; ++it) {
    const int bound = bounds[it];
    v8f acc[2] = {};

#pragma unroll
    for (int s = 0; s < 2; ++s) {
      const int t = wv + s * 16;                     // row tile (16 rows)
      if (t >= 25 || t * 16 >= bound) continue;      // wave-uniform branch
      const _Float16* arow = Ml + (t * 16 + rsub) * SM;
      const _Float16* hb0  = hl + (hi ? 8 : 0);
      v8f c = {};
      for (int kt = 0; kt < 12; ++kt) {              // full K tiles 0..383
        const int kb = kt * 32;
        union { v16h v; h4 q[4]; } A, Bm;
        A.q[0] = *(const h4*)(arow + kb + lo);
        A.q[1] = *(const h4*)(arow + kb + lo + 4);
        A.q[2] = *(const h4*)(arow + kb + lo + 16);
        A.q[3] = *(const h4*)(arow + kb + lo + 20);
        Bm.q[0] = *(const h4*)(hb0 + kb);            // same addr lanes<16 -> bcast
        Bm.q[1] = *(const h4*)(hb0 + kb + 4);
        Bm.q[2] = *(const h4*)(hb0 + kb + 16);
        Bm.q[3] = *(const h4*)(hb0 + kb + 20);
        c = __builtin_amdgcn_wmma_f32_16x16x32_f16(
                false, A.v, false, Bm.v, (short)0, c, false, false);
      }
      {                                              // K tail 384..415
        const int kb = 384;
        union { v16h v; h4 q[4]; } A, Bm;
        A.q[0] = *(const h4*)(arow + kb + lo);       // cols 384..399 valid
        A.q[1] = *(const h4*)(arow + kb + lo + 4);
        A.q[2] = hz;                                 // cols 400..415 -> 0
        A.q[3] = hz;
        Bm.q[0] = *(const h4*)(hb0 + kb);            // hl zero-padded to 416
        Bm.q[1] = *(const h4*)(hb0 + kb + 4);
        Bm.q[2] = *(const h4*)(hb0 + kb + 16);
        Bm.q[3] = *(const h4*)(hb0 + kb + 20);
        c = __builtin_amdgcn_wmma_f32_16x16x32_f16(
                false, A.v, false, Bm.v, (short)0, c, false, false);
      }
      acc[s] = c;
    }
    __syncthreads();   // all reads of h done before anyone rewrites it

    // D layout: lane 0 holds rows t*16+0..7 in c[0..7]; lane 16 rows +8..15
    if ((lane & 15) == 0) {
#pragma unroll
      for (int s = 0; s < 2; ++s) {
        const int t = wv + s * 16;
        if (t >= 25 || t * 16 >= bound) continue;
        const int rbase = t * 16 + (hi ? 8 : 0);
#pragma unroll
        for (int r = 0; r < 8; ++r) {
          const int row = rbase + r;
          if (row < bound) {
            float u = 0.8f * hf[row] + acc[s][r];
            u = lrelu(clip1(u));
            hf[row] = u;
            hl[row] = (_Float16)u;
          }
        }
      }
    }
    __syncthreads();
  }

  // ---------- Output --------------------------------------------------------
  for (int i = tid; i < P; i += NT) out[(size_t)b * P + i] = hf[i];
}

extern "C" void kernel_launch(void* const* d_in, const int* in_sizes, int n_in,
                              void* d_out, int out_size, void* d_ws, size_t ws_size,
                              hipStream_t stream) {
  (void)n_in; (void)out_size; (void)d_ws; (void)ws_size;
  const float* M_prev = (const float*)d_in[0];
  const float* p_inf  = (const float*)d_in[1];
  const float* p_gen  = (const float*)d_in[2];
  const float* query  = (const float*)d_in[3];
  float* out = (float*)d_out;
  const int B = in_sizes[1] / P;   // 512
  hipLaunchKernelGGL(HierarchicalMemory_37838661878388_kernel,
                     dim3(B), dim3(NT), LDS_BYTES, stream,
                     M_prev, p_inf, p_gen, query, out);
}
